// LSTM1997Layer_45423574122820
// MI455X (gfx1250) — matrix operand order repaired and checked by reading, not compiled, not run on hardware
//
#include <hip/hip_runtime.h>
#include <hip/hip_bf16.h>

#define B_    64
#define S_    512
#define IN_   512
#define NBLK_ 64
#define DBLK_ 16
#define DHID_ 1024
#define KTOT_ (IN_ + DHID_)   // 1536

typedef __attribute__((ext_vector_type(16))) __bf16 v16bf;
typedef __attribute__((ext_vector_type(8)))  __bf16 bf16x8;
typedef __attribute__((ext_vector_type(8)))  float  v8f;

__device__ __forceinline__ __bf16 f2bf(float f) {
  unsigned u = __float_as_uint(f);
  unsigned short s = (unsigned short)((u + 0x7FFFu + ((u >> 16) & 1u)) >> 16);
  __bf16 r;
  __builtin_memcpy(&r, &s, sizeof(r));
  return r;
}

__device__ __forceinline__ float sigmoidf_(float x) {
  return 1.0f / (1.0f + __expf(-x));
}

// ---------------- Phase 0a: x (B,S,IN) f32 -> xb (S,B,IN) bf16 ----------------
__global__ void k_convert_x(const float* __restrict__ x, __bf16* __restrict__ xb) {
  size_t e8 = ((size_t)blockIdx.x * blockDim.x + threadIdx.x) * 8;
  if (e8 >= (size_t)S_ * B_ * IN_) return;
  int k = (int)(e8 % IN_);
  int r = (int)((e8 / IN_) % B_);
  int t = (int)(e8 / ((size_t)IN_ * B_));
  const float* src = x + ((size_t)r * S_ + t) * IN_ + k;
  __bf16* dst = xb + e8;
#pragma unroll
  for (int i = 0; i < 8; ++i) dst[i] = f2bf(src[i]);
}

// ---------------- Phase 0b: build transposed bf16 weights ----------------
// WmcT[n][k] = (k<IN ? Wmc[k][n] : Umc[k-IN][n]),   shape (DHID, KTOT)
// GW2[b][g][k] = gate column b of (g==0 ? [Wig;Uig] : [Wog;Uog]), shape (NBLK,2,KTOT)
__global__ void k_build_w(const float* __restrict__ Wmc, const float* __restrict__ Umc,
                          const float* __restrict__ Wig, const float* __restrict__ Uig,
                          const float* __restrict__ Wog, const float* __restrict__ Uog,
                          __bf16* __restrict__ WmcT, __bf16* __restrict__ GW2) {
  int idx = blockIdx.x * blockDim.x + threadIdx.x;
  const int NW = DHID_ * KTOT_;
  if (idx < NW) {
    int k = idx % KTOT_;
    int n = idx / KTOT_;
    float v = (k < IN_) ? Wmc[(size_t)k * DHID_ + n] : Umc[(size_t)(k - IN_) * DHID_ + n];
    WmcT[idx] = f2bf(v);
  } else {
    int j = idx - NW;
    if (j < NBLK_ * 2 * KTOT_) {
      int k  = j % KTOT_;
      int cb = j / KTOT_;      // b*2 + g
      int b  = cb >> 1;
      int g  = cb & 1;
      const float* W = g ? Wog : Wig;
      const float* U = g ? Uog : Uig;
      float v = (k < IN_) ? W[(size_t)k * NBLK_ + b] : U[(size_t)(k - IN_) * NBLK_ + b];
      GW2[j] = f2bf(v);
    }
  }
}

// ---------------- Phase 0c: init h buffer (parity 0) + zero barrier ----------------
__global__ void k_init_h(const float* __restrict__ h0, __bf16* __restrict__ hbuf,
                         unsigned* __restrict__ bar) {
  int idx = blockIdx.x * blockDim.x + threadIdx.x;
  if (idx < B_ * DHID_) hbuf[idx] = f2bf(h0[idx % DHID_]);
  if (idx < 2) bar[idx] = 0u;
}

// ---------------- Persistent scan kernel: 64 WGs (one per block) x 128 threads ----------------
__global__ __launch_bounds__(128)
void lstm97_scan_kernel(const __bf16* __restrict__ xb,      // (S,B,IN) bf16
                        const __bf16* __restrict__ WmcT,    // (DHID,KTOT) bf16
                        const __bf16* __restrict__ GW2,     // (NBLK,2,KTOT) bf16
                        __bf16* __restrict__ hbuf,          // 2 x (B,DHID) bf16 double buffer
                        unsigned* __restrict__ bar,         // [0]=cnt [1]=epoch
                        const float* __restrict__ big, const float* __restrict__ bog,
                        const float* __restrict__ bmc, const float* __restrict__ c0,
                        float* __restrict__ out) {
  __shared__ __align__(32) __bf16 sW[DBLK_ * KTOT_];   // 48 KB: this block's 16 WmcT rows ([Wmc;Umc]^T)
  __shared__ __align__(32) __bf16 sG[3 * KTOT_];       // 9 KB: ig col, og col, zero col
  __shared__ float sGO[B_ * 2];                        // per-row ig/og pre-activations

  const int b    = blockIdx.x;
  const int tid  = threadIdx.x;
  const int lane = tid & 31;
  const int wave = tid >> 5;       // M-tile 0..3
  const int m0   = wave * 16;
  const int col  = lane & 15;      // B/D column, also A row-within-tile
  const int sel  = lane >> 4;
  const int colz = (col < 2) ? col : 2;   // gate B column (2 -> zero column)

  // Stage step-invariant B operands into LDS (contiguous copies).
  for (int i = tid; i < DBLK_ * KTOT_; i += 128)
    sW[i] = WmcT[(size_t)(b * DBLK_) * KTOT_ + i];
  for (int i = tid; i < 2 * KTOT_; i += 128)
    sG[i] = GW2[(size_t)(b * 2) * KTOT_ + i];
  for (int i = tid; i < KTOT_; i += 128)          // zero column
    sG[2 * KTOT_ + i] = f2bf(0.0f);
  __syncthreads();

  const float bmc_l = bmc[b * DBLK_ + col];
  const float big_b = big[b];
  const float bog_b = bog[b];

  // Cell state lives in accumulator layout: element (row = m0+v+8*sel, col).
  v8f c;
  {
    float cv = c0[b * DBLK_ + col];    // broadcast over batch rows
#pragma unroll
    for (int v = 0; v < 8; ++v) c[v] = cv;
  }

  float* hseq = out;
  float* cseq = out + (size_t)B_ * S_ * DHID_;

  const int arow = m0 + col;  // A-fragment batch row for this lane
  const __bf16* sWc = sW + col * KTOT_ + sel * 16;
  const __bf16* sGc = sG + colz * KTOT_ + sel * 16;

  for (int t = 0; t < S_; ++t) {
    const __bf16* xt = xb + (size_t)t * B_ * IN_;
    const __bf16* hB = hbuf + (size_t)(t & 1) * B_ * DHID_;
    __bf16*       hN = hbuf + (size_t)((t + 1) & 1) * B_ * DHID_;

    if (t + 1 < S_) {  // prefetch next step's x tile (64 KB across 128 threads)
      const char* nxt = (const char*)(xb + (size_t)(t + 1) * B_ * IN_) + tid * 512;
      __builtin_prefetch(nxt, 0, 1);
    }

    const __bf16* arow_x = xt + (size_t)arow * IN_ + sel * 16;
    const __bf16* arow_h = hB + (size_t)arow * DHID_ + sel * 16;

    v8f acc = {0.f, 0.f, 0.f, 0.f, 0.f, 0.f, 0.f, 0.f};
    v8f gac = {0.f, 0.f, 0.f, 0.f, 0.f, 0.f, 0.f, 0.f};

    // ---- K over x part (k = 0..511) ----
#pragma unroll 8
    for (int kc = 0; kc < 16; ++kc) {
      const int k0 = kc * 32;
      bf16x8 a0 = *(const bf16x8*)(arow_x + k0);
      bf16x8 a1 = *(const bf16x8*)(arow_x + k0 + 8);
      v16bf A = __builtin_shufflevector(a0, a1, 0,1,2,3,4,5,6,7,8,9,10,11,12,13,14,15);
      v16bf Bm = *(const v16bf*)(sWc + k0);
      v16bf Bg = *(const v16bf*)(sGc + k0);
      acc = __builtin_amdgcn_wmma_f32_16x16x32_bf16(false, A, false, Bm, (short)0, acc, false, false);
      gac = __builtin_amdgcn_wmma_f32_16x16x32_bf16(false, A, false, Bg, (short)0, gac, false, false);
    }
    // ---- K over h part (k = 512..1535) ----
#pragma unroll 8
    for (int kc = 0; kc < 32; ++kc) {
      const int k0 = kc * 32;
      const int kb = IN_ + k0;
      bf16x8 a0 = *(const bf16x8*)(arow_h + k0);
      bf16x8 a1 = *(const bf16x8*)(arow_h + k0 + 8);
      v16bf A = __builtin_shufflevector(a0, a1, 0,1,2,3,4,5,6,7,8,9,10,11,12,13,14,15);
      v16bf Bm = *(const v16bf*)(sWc + kb);
      v16bf Bg = *(const v16bf*)(sGc + kb);
      acc = __builtin_amdgcn_wmma_f32_16x16x32_bf16(false, A, false, Bm, (short)0, acc, false, false);
      gac = __builtin_amdgcn_wmma_f32_16x16x32_bf16(false, A, false, Bg, (short)0, gac, false, false);
    }

    // Publish gate pre-activations (cols 0/1 of gate tile) to LDS.
    if (col < 2) {
#pragma unroll
      for (int v = 0; v < 8; ++v) {
        int r = m0 + v + 8 * sel;
        sGO[r * 2 + col] = gac[v];
      }
    }
    __syncthreads();

    // Epilogue: gates, CEC update, outputs, next-h.
#pragma unroll
    for (int v = 0; v < 8; ++v) {
      int r = m0 + v + 8 * sel;
      float g  = tanhf(acc[v] + bmc_l);
      float ig = sigmoidf_(sGO[r * 2 + 0] + big_b);
      float og = sigmoidf_(sGO[r * 2 + 1] + bog_b);
      float cn = c[v] + ig * g;
      float hn = og * tanhf(cn);
      c[v] = cn;
      size_t o = ((size_t)r * S_ + t) * DHID_ + (size_t)b * DBLK_ + col;
      hseq[o] = hn;
      cseq[o] = cn;
      hN[(size_t)r * DHID_ + (size_t)b * DBLK_ + col] = f2bf(hn);
    }

    // Device-wide sense barrier (epoch target = t+1).
    __threadfence();
    __syncthreads();
    if (tid == 0) {
      unsigned old = __hip_atomic_fetch_add(&bar[0], 1u, __ATOMIC_ACQ_REL, __HIP_MEMORY_SCOPE_AGENT);
      if (old == (unsigned)(NBLK_ - 1)) {
        __hip_atomic_store(&bar[0], 0u, __ATOMIC_RELAXED, __HIP_MEMORY_SCOPE_AGENT);
        __hip_atomic_fetch_add(&bar[1], 1u, __ATOMIC_RELEASE, __HIP_MEMORY_SCOPE_AGENT);
      } else {
        while (__hip_atomic_load(&bar[1], __ATOMIC_ACQUIRE, __HIP_MEMORY_SCOPE_AGENT) <
               (unsigned)(t + 1))
          __builtin_amdgcn_s_sleep(1);
      }
    }
    __syncthreads();
    __threadfence();
  }
}

extern "C" void kernel_launch(void* const* d_in, const int* in_sizes, int n_in,
                              void* d_out, int out_size, void* d_ws, size_t ws_size,
                              hipStream_t stream) {
  const float* x   = (const float*)d_in[0];
  const float* h0  = (const float*)d_in[1];
  const float* c0  = (const float*)d_in[2];
  const float* Wig = (const float*)d_in[3];
  const float* Uig = (const float*)d_in[4];
  const float* big = (const float*)d_in[5];
  const float* Wog = (const float*)d_in[6];
  const float* Uog = (const float*)d_in[7];
  const float* bog = (const float*)d_in[8];
  const float* Wmc = (const float*)d_in[9];
  const float* Umc = (const float*)d_in[10];
  const float* bmc = (const float*)d_in[11];

  char* ws = (char*)d_ws;
  size_t off = 0;
  auto alloc = [&](size_t bytes) -> void* {
    void* p = ws + off;
    off = (off + bytes + 255) & ~(size_t)255;
    return p;
  };
  __bf16*   xb   = (__bf16*)alloc((size_t)S_ * B_ * IN_ * 2);        // 32 MB
  __bf16*   WmcT = (__bf16*)alloc((size_t)DHID_ * KTOT_ * 2);        // 3 MB
  __bf16*   GW2  = (__bf16*)alloc((size_t)NBLK_ * 2 * KTOT_ * 2);    // 384 KB
  __bf16*   hbuf = (__bf16*)alloc((size_t)2 * B_ * DHID_ * 2);       // 256 KB
  unsigned* bar  = (unsigned*)alloc(256);

  {
    size_t n8 = (size_t)S_ * B_ * IN_ / 8;
    k_convert_x<<<(unsigned)((n8 + 255) / 256), 256, 0, stream>>>(x, xb);
  }
  {
    int n = DHID_ * KTOT_ + NBLK_ * 2 * KTOT_;
    k_build_w<<<(n + 255) / 256, 256, 0, stream>>>(Wmc, Umc, Wig, Uig, Wog, Uog, WmcT, GW2);
  }
  k_init_h<<<(B_ * DHID_ + 255) / 256, 256, 0, stream>>>(h0, hbuf, bar);

  lstm97_scan_kernel<<<NBLK_, 128, 0, stream>>>(xb, WmcT, GW2, hbuf, bar,
                                                big, bog, bmc, c0, (float*)d_out);
}
